// InvariantGraphEncoder_29858612642365
// MI455X (gfx1250) — matrix operand order repaired and checked by reading, not compiled, not run on hardware
//
#include <hip/hip_runtime.h>
#include <hip/hip_bf16.h>

typedef unsigned short u16;
typedef __bf16 v16bf __attribute__((ext_vector_type(16)));
typedef float  v8f   __attribute__((ext_vector_type(8)));

#define N_NODES 50000
#define N_EDGES 800000
#define IN_NF   32
#define HID     64
#define ENF     5

// ---------- small helpers ----------
__device__ inline u16 f2bf(float f) {            // RNE f32 -> bf16 bits
  unsigned u = __float_as_uint(f);
  unsigned r = (u + 0x7FFFu + ((u >> 16) & 1u)) >> 16;
  return (u16)r;
}
// SiLU with hardware transcendentals: v_exp_f32 + v_rcp_f32 (co-issue with WMMA)
__device__ inline float silu(float y) {
  return y * __builtin_amdgcn_rcpf(1.0f + __expf(-y));
}

struct B32 { uint4 a, b; };
__device__ inline v16bf mk16(uint4 a, uint4 b) {
  B32 p{a, b};
  return __builtin_bit_cast(v16bf, p);
}

// A-fragment (16xK tile, bf16, row-major in LDS, `stride` halves per row), K-chunk kc.
// Lane layout per ISA 7.12.2: lanes<16 hold K {0..7,16..23}+kc*32, lanes>=16 hold {8..15,24..31}.
__device__ inline v16bf ldsA(const u16* A, int stride, int kc, int lane) {
  int m  = lane & 15;
  int ho = (lane & 16) ? 8 : 0;
  const uint4* p = (const uint4*)(A + m * stride + kc * 32 + ho);
  return mk16(p[0], p[2]);   // +0 and +16 halves
}
// B-fragment pre-packed in global memory: 16 contiguous halves per lane.
__device__ inline v16bf glbB(const u16* wf, int kc, int nt, int lane) {
  const uint4* p = (const uint4*)(wf + ((((kc << 2) + nt) << 5) + lane) * 16);
  return mk16(p[0], p[1]);
}
__device__ inline v8f wmma(v16bf a, v16bf b, v8f c) {
  return __builtin_amdgcn_wmma_f32_16x16x32_bf16(false, a, false, b, (short)0, c, false, false);
}
__device__ inline void init_bias(v8f acc[4], const float* b, int lane) {
  int nlo = lane & 15;
#pragma unroll
  for (int nt = 0; nt < 4; ++nt) {
    float v = b[nt * 16 + nlo];
#pragma unroll
    for (int i = 0; i < 8; ++i) acc[nt][i] = v;
  }
}
__device__ inline void gemm_acc(v8f acc[4], const u16* A, int astride, const u16* wf, int KC, int lane) {
  for (int kc = 0; kc < KC; ++kc) {
    v16bf a = ldsA(A, astride, kc, lane);
#pragma unroll
    for (int nt = 0; nt < 4; ++nt) acc[nt] = wmma(a, glbB(wf, kc, nt, lane), acc[nt]);
  }
}
// LayerNorm(64) + SiLU on C-layout accumulators, store bf16 into LDS (A-tile for 2nd GEMM).
__device__ inline void ln_silu_store(v8f acc[4], const float* g64, const float* b64,
                                     u16* t2, int ts, int lane) {
  int nlo = lane & 15, mo = (lane & 16) ? 8 : 0;
  float gg[4], bb[4];
#pragma unroll
  for (int nt = 0; nt < 4; ++nt) { gg[nt] = g64[nt * 16 + nlo]; bb[nt] = b64[nt * 16 + nlo]; }
#pragma unroll
  for (int g = 0; g < 8; ++g) {
    float s = acc[0][g] + acc[1][g] + acc[2][g] + acc[3][g];
    float q = acc[0][g]*acc[0][g] + acc[1][g]*acc[1][g] + acc[2][g]*acc[2][g] + acc[3][g]*acc[3][g];
#pragma unroll
    for (int msk = 1; msk < 16; msk <<= 1) { s += __shfl_xor(s, msk, 32); q += __shfl_xor(q, msk, 32); }
    float mean = s * (1.f / 64.f);
    float var  = q * (1.f / 64.f) - mean * mean;
    float rs   = rsqrtf(var + 1e-5f);
    int M = g + mo;
#pragma unroll
    for (int nt = 0; nt < 4; ++nt) {
      float y = (acc[nt][g] - mean) * rs * gg[nt] + bb[nt];
      t2[M * ts + nt * 16 + nlo] = f2bf(silu(y));
    }
  }
}
__device__ inline void silu_store(v8f acc[4], u16* t2, int ts, int lane) {
  int nlo = lane & 15, mo = (lane & 16) ? 8 : 0;
#pragma unroll
  for (int g = 0; g < 8; ++g) {
    int M = g + mo;
#pragma unroll
    for (int nt = 0; nt < 4; ++nt) t2[M * ts + nt * 16 + nlo] = f2bf(silu(acc[nt][g]));
  }
}

// ---------- weight pre-pack: f32 [K x 64] row-major -> bf16 B-fragment layout ----------
__global__ void k_pack_w(const float* __restrict__ w, u16* __restrict__ wf, int Kact, int KC) {
  int tid = blockIdx.x * blockDim.x + threadIdx.x;
  int total = KC * 4 * 32 * 16;
  if (tid >= total) return;
  int j = tid & 15, lane = (tid >> 4) & 31, f = tid >> 9;
  int kc = f >> 2, nt = f & 3;
  int k = kc * 32 + ((lane < 16) ? 0 : 8) + ((j < 8) ? j : j + 8);
  int n = nt * 16 + (lane & 15);
  wf[tid] = (k < Kact) ? f2bf(w[k * 64 + n]) : (u16)0;
}

// ---------- embedding_in: SiLU(h@W1+b1)@W2+b2 ----------
__global__ __launch_bounds__(256) void k_embed_in(const float* __restrict__ h,
    const u16* __restrict__ w1f, const float* __restrict__ b1,
    const u16* __restrict__ w2f, const float* __restrict__ b2,
    float* __restrict__ x, int ntiles) {
  __shared__ __align__(16) u16 Ash[8][16 * 40];
  __shared__ __align__(16) u16 T2[8][16 * 72];
  int wv = threadIdx.x >> 5, lane = threadIdx.x & 31;
  int tile = blockIdx.x * 8 + wv;
  bool valid = tile < ntiles;
  int base = (valid ? tile : 0) * 16;
  int m = lane & 15, ch = lane >> 4;
  {
    const float4* src = (const float4*)(h + (base + m) * IN_NF + ch * 16);
    u16* dst = &Ash[wv][m * 40 + ch * 16];
#pragma unroll
    for (int q = 0; q < 4; ++q) {
      float4 v = src[q];
      dst[q*4+0]=f2bf(v.x); dst[q*4+1]=f2bf(v.y); dst[q*4+2]=f2bf(v.z); dst[q*4+3]=f2bf(v.w);
    }
  }
  __syncthreads();
  v8f acc[4]; init_bias(acc, b1, lane);
  gemm_acc(acc, Ash[wv], 40, w1f, 1, lane);
  silu_store(acc, T2[wv], 72, lane);
  __syncthreads();
  v8f a2[4]; init_bias(a2, b2, lane);
  gemm_acc(a2, T2[wv], 72, w2f, 2, lane);
  if (valid) {
    int nlo = lane & 15, mo = (lane & 16) ? 8 : 0;
#pragma unroll
    for (int g = 0; g < 8; ++g)
#pragma unroll
      for (int nt = 0; nt < 4; ++nt)
        x[(base + g + mo) * HID + nt * 16 + nlo] = a2[nt][g];
  }
}

// ---------- LayerNorm + bf16 snapshot + agg zeroing (one node per wave) ----------
__global__ __launch_bounds__(256) void k_ln(const float* __restrict__ x,
    const float* __restrict__ ng, const float* __restrict__ nb,
    float* __restrict__ hn, u16* __restrict__ hnb, float* __restrict__ agg) {
  int wv = threadIdx.x >> 5, lane = threadIdx.x & 31;
  int node = blockIdx.x * 8 + wv;
  const float* xr = x + node * HID;
  float v0 = xr[lane], v1 = xr[lane + 32];
  float s = v0 + v1, q = v0 * v0 + v1 * v1;
#pragma unroll
  for (int msk = 1; msk < 32; msk <<= 1) { s += __shfl_xor(s, msk, 32); q += __shfl_xor(q, msk, 32); }
  float mean = s * (1.f / 64.f), var = q * (1.f / 64.f) - mean * mean, rs = rsqrtf(var + 1e-5f);
  float y0 = (v0 - mean) * rs * ng[lane]      + nb[lane];
  float y1 = (v1 - mean) * rs * ng[lane + 32] + nb[lane + 32];
  int o = node * HID + lane;
  hn[o] = y0; hn[o + 32] = y1;
  hnb[o] = f2bf(y0); hnb[o + 32] = f2bf(y1);
  agg[o] = 0.f; agg[o + 32] = 0.f;
}

// ---------- edge MLP + atomic scatter-add (16 edges per wave) ----------
__global__ __launch_bounds__(256) void k_edge(const u16* __restrict__ hnb,
    const int* __restrict__ row, const int* __restrict__ col, const float* __restrict__ ea,
    const u16* __restrict__ w1f, const float* __restrict__ b1,
    const float* __restrict__ lg, const float* __restrict__ lb,
    const u16* __restrict__ w2f, const float* __restrict__ b2,
    float* __restrict__ agg) {
  __shared__ __align__(16) u16 Ash[8][16 * 168];   // 16 x 160 (pad->168) bf16 input tile
  __shared__ __align__(16) u16 T2[8][16 * 72];
  __shared__ int RIDX[8][16];
  int wv = threadIdx.x >> 5, lane = threadIdx.x & 31;
  int eb = (blockIdx.x * 8 + wv) * 16;
  int m = lane & 15;
  u16* Aw = Ash[wv];
  int e = eb + m;
  if (lane < 16) {                                  // hn[row], edge_attr, zero-pad
    int r = row[e];
    RIDX[wv][m] = r;
    const uint4* src = (const uint4*)(hnb + r * HID);
    uint4* dst = (uint4*)(Aw + m * 168);
    dst[0] = src[0]; dst[1] = src[1]; dst[2] = src[2]; dst[3] = src[3];
    const float* ep = ea + e * ENF;
#pragma unroll
    for (int j = 0; j < ENF; ++j) Aw[m * 168 + 128 + j] = f2bf(ep[j]);
#pragma unroll
    for (int j = ENF; j < 32; ++j) Aw[m * 168 + 128 + j] = 0;
  } else {                                          // hn[col]
    int c = col[e];
    const uint4* src = (const uint4*)(hnb + c * HID);
    uint4* dst = (uint4*)(Aw + m * 168 + 32);       // +32 halves
    dst[0] = src[0]; dst[1] = src[1]; dst[2] = src[2]; dst[3] = src[3];
  }
  __syncthreads();
  v8f acc[4]; init_bias(acc, b1, lane);
  gemm_acc(acc, Aw, 168, w1f, 5, lane);             // K = 160 (133 padded)
  ln_silu_store(acc, lg, lb, T2[wv], 72, lane);
  __syncthreads();
  v8f a2[4]; init_bias(a2, b2, lane);
  gemm_acc(a2, T2[wv], 72, w2f, 2, lane);
  int nlo = lane & 15, mo = (lane & 16) ? 8 : 0;
#pragma unroll
  for (int g = 0; g < 8; ++g) {
    int r = RIDX[wv][g + mo];
    float* ap = agg + r * HID + nlo;
#pragma unroll
    for (int nt = 0; nt < 4; ++nt) atomicAdd(ap + nt * 16, a2[nt][g]);
  }
}

// ---------- node MLP + residual (16 nodes per wave) ----------
__global__ __launch_bounds__(256) void k_node(const float* __restrict__ hn,
    const u16* __restrict__ hnb, const float* __restrict__ agg,
    const u16* __restrict__ w1f, const float* __restrict__ b1,
    const float* __restrict__ lg, const float* __restrict__ lb,
    const u16* __restrict__ w2f, const float* __restrict__ b2,
    float* __restrict__ xo, int ntiles) {
  __shared__ __align__(16) u16 Ash[8][16 * 136];   // 16 x 128 (pad->136)
  __shared__ __align__(16) u16 T2[8][16 * 72];
  int wv = threadIdx.x >> 5, lane = threadIdx.x & 31;
  int tile = blockIdx.x * 8 + wv;
  bool valid = tile < ntiles;
  int base = (valid ? tile : 0) * 16;
  int m = lane & 15;
  u16* Aw = Ash[wv];
  if (lane < 16) {                                  // hn (already bf16)
    const uint4* src = (const uint4*)(hnb + (base + m) * HID);
    uint4* dst = (uint4*)(Aw + m * 136);
    dst[0] = src[0]; dst[1] = src[1]; dst[2] = src[2]; dst[3] = src[3];
  } else {                                          // agg f32 -> bf16
    const float4* src = (const float4*)(agg + (base + m) * HID);
    u16* d = Aw + m * 136 + 64;
#pragma unroll
    for (int q = 0; q < 16; ++q) {
      float4 v = src[q];
      d[q*4+0]=f2bf(v.x); d[q*4+1]=f2bf(v.y); d[q*4+2]=f2bf(v.z); d[q*4+3]=f2bf(v.w);
    }
  }
  __syncthreads();
  v8f acc[4]; init_bias(acc, b1, lane);
  gemm_acc(acc, Aw, 136, w1f, 4, lane);             // K = 128
  ln_silu_store(acc, lg, lb, T2[wv], 72, lane);
  __syncthreads();
  v8f a2[4]; init_bias(a2, b2, lane);
  gemm_acc(a2, T2[wv], 72, w2f, 2, lane);
  if (valid) {
    int nlo = lane & 15, mo = (lane & 16) ? 8 : 0;
#pragma unroll
    for (int g = 0; g < 8; ++g)
#pragma unroll
      for (int nt = 0; nt < 4; ++nt) {
        int idx = (base + g + mo) * HID + nt * 16 + nlo;
        xo[idx] = hn[idx] + a2[nt][g];              // residual on normed h
      }
  }
}

// ---------- embedding_out ----------
__global__ __launch_bounds__(256) void k_embed_out(const float* __restrict__ x,
    const u16* __restrict__ w1f, const float* __restrict__ b1,
    const u16* __restrict__ w2f, const float* __restrict__ b2,
    float* __restrict__ out, int ntiles) {
  __shared__ __align__(16) u16 Ash[8][16 * 72];
  __shared__ __align__(16) u16 T2[8][16 * 72];
  int wv = threadIdx.x >> 5, lane = threadIdx.x & 31;
  int tile = blockIdx.x * 8 + wv;
  bool valid = tile < ntiles;
  int base = (valid ? tile : 0) * 16;
  int m = lane & 15, ch = lane >> 4;
  {
    const float4* src = (const float4*)(x + (base + m) * HID + ch * 32);
    u16* dst = &Ash[wv][m * 72 + ch * 32];
#pragma unroll
    for (int q = 0; q < 8; ++q) {
      float4 v = src[q];
      dst[q*4+0]=f2bf(v.x); dst[q*4+1]=f2bf(v.y); dst[q*4+2]=f2bf(v.z); dst[q*4+3]=f2bf(v.w);
    }
  }
  __syncthreads();
  v8f acc[4]; init_bias(acc, b1, lane);
  gemm_acc(acc, Ash[wv], 72, w1f, 2, lane);
  silu_store(acc, T2[wv], 72, lane);
  __syncthreads();
  v8f a2[4]; init_bias(a2, b2, lane);
  gemm_acc(a2, T2[wv], 72, w2f, 2, lane);
  if (valid) {
    int nlo = lane & 15, mo = (lane & 16) ? 8 : 0;
#pragma unroll
    for (int g = 0; g < 8; ++g)
#pragma unroll
      for (int nt = 0; nt < 4; ++nt)
        out[(base + g + mo) * HID + nt * 16 + nlo] = a2[nt][g];
  }
}

// ---------- host orchestration ----------
extern "C" void kernel_launch(void* const* d_in, const int* in_sizes, int n_in,
                              void* d_out, int out_size, void* d_ws, size_t ws_size,
                              hipStream_t stream) {
  (void)in_sizes; (void)n_in; (void)out_size; (void)ws_size;
  const float* h    = (const float*)d_in[0];
  const int*   edg  = (const int*)d_in[1];
  const int*   row  = edg;
  const int*   col  = edg + N_EDGES;
  const float* ea   = (const float*)d_in[2];
  // params flattened as pytree leaves (sorted dict keys)
  int p = 3;
  const float* ei_b1 = (const float*)d_in[p++];
  const float* ei_b2 = (const float*)d_in[p++];
  const float* ei_w1 = (const float*)d_in[p++];
  const float* ei_w2 = (const float*)d_in[p++];
  const float* eo_b1 = (const float*)d_in[p++];
  const float* eo_b2 = (const float*)d_in[p++];
  const float* eo_w1 = (const float*)d_in[p++];
  const float* eo_w2 = (const float*)d_in[p++];
  struct Layer {
    const float *e_b1,*e_b2,*e_be,*e_g,*e_w1,*e_w2,*nb,*ng,*n_b1,*n_b2,*n_be,*n_g,*n_w1,*n_w2;
  } L[2];
  for (int i = 0; i < 2; ++i) {
    L[i].e_b1=(const float*)d_in[p++]; L[i].e_b2=(const float*)d_in[p++];
    L[i].e_be=(const float*)d_in[p++]; L[i].e_g =(const float*)d_in[p++];
    L[i].e_w1=(const float*)d_in[p++]; L[i].e_w2=(const float*)d_in[p++];
    L[i].nb  =(const float*)d_in[p++]; L[i].ng  =(const float*)d_in[p++];
    L[i].n_b1=(const float*)d_in[p++]; L[i].n_b2=(const float*)d_in[p++];
    L[i].n_be=(const float*)d_in[p++]; L[i].n_g =(const float*)d_in[p++];
    L[i].n_w1=(const float*)d_in[p++]; L[i].n_w2=(const float*)d_in[p++];
  }

  // workspace carve-out
  char* ws = (char*)d_ws;
  size_t cur = 0;
  auto alloc = [&](size_t bytes) { void* r = ws + cur; cur = (cur + bytes + 255) & ~(size_t)255; return r; };
  float* x    = (float*)alloc((size_t)N_NODES * HID * 4);
  float* hn   = (float*)alloc((size_t)N_NODES * HID * 4);
  u16*   hnb  = (u16*)  alloc((size_t)N_NODES * HID * 2);
  float* agg  = (float*)alloc((size_t)N_NODES * HID * 4);
  auto wsz = [](int KC) { return (size_t)KC * 4 * 32 * 16 * 2; };
  u16* ei_w1f = (u16*)alloc(wsz(1));
  u16* ei_w2f = (u16*)alloc(wsz(2));
  u16* eo_w1f = (u16*)alloc(wsz(2));
  u16* eo_w2f = (u16*)alloc(wsz(2));
  u16 *ew1f[2], *ew2f[2], *nw1f[2], *nw2f[2];
  for (int i = 0; i < 2; ++i) {
    ew1f[i] = (u16*)alloc(wsz(5));
    ew2f[i] = (u16*)alloc(wsz(2));
    nw1f[i] = (u16*)alloc(wsz(4));
    nw2f[i] = (u16*)alloc(wsz(2));
  }

  auto pack = [&](const float* w, u16* wf, int Kact, int KC) {
    int total = KC * 4 * 32 * 16;
    k_pack_w<<<(total + 255) / 256, 256, 0, stream>>>(w, wf, Kact, KC);
  };
  pack(ei_w1, ei_w1f, 32, 1);
  pack(ei_w2, ei_w2f, 64, 2);
  pack(eo_w1, eo_w1f, 64, 2);
  pack(eo_w2, eo_w2f, 64, 2);
  for (int i = 0; i < 2; ++i) {
    pack(L[i].e_w1, ew1f[i], 2 * HID + ENF, 5);   // 133 -> 160
    pack(L[i].e_w2, ew2f[i], 64, 2);
    pack(L[i].n_w1, nw1f[i], 128, 4);
    pack(L[i].n_w2, nw2f[i], 64, 2);
  }

  const int nodeTiles = N_NODES / 16;              // 3125
  const int nodeBlocks = (nodeTiles + 7) / 8;      // 391
  const int edgeBlocks = N_EDGES / 16 / 8;         // 6250
  const int lnBlocks   = N_NODES / 8;              // 6250

  k_embed_in<<<nodeBlocks, 256, 0, stream>>>(h, ei_w1f, ei_b1, ei_w2f, ei_b2, x, nodeTiles);
  for (int i = 0; i < 2; ++i) {
    k_ln<<<lnBlocks, 256, 0, stream>>>(x, L[i].ng, L[i].nb, hn, hnb, agg);
    k_edge<<<edgeBlocks, 256, 0, stream>>>(hnb, row, col, ea,
        ew1f[i], L[i].e_b1, L[i].e_g, L[i].e_be, ew2f[i], L[i].e_b2, agg);
    k_node<<<nodeBlocks, 256, 0, stream>>>(hn, hnb, agg,
        nw1f[i], L[i].n_b1, L[i].n_g, L[i].n_be, nw2f[i], L[i].n_b2, x, nodeTiles);
  }
  k_embed_out<<<nodeBlocks, 256, 0, stream>>>(x, eo_w1f, eo_b1, eo_w2f, eo_b2, (float*)d_out, nodeTiles);
}
